// LengthRegulator_21534966022208
// MI455X (gfx1250) — matrix operand order repaired and checked by reading, not compiled, not run on hardware
//
#include <hip/hip_runtime.h>
#include <hip/hip_bf16.h>

typedef __attribute__((ext_vector_type(16))) __bf16          v16bf;
typedef __attribute__((ext_vector_type(8)))  float           v8f;
typedef __attribute__((ext_vector_type(8)))  unsigned short  us8;

#define B_ 32
#define N_ 512
#define D_ 384
#define F_ 256
#define M_ 3072
#define KC 32           // K per WMMA chunk
#define TILE_ROWS 64

// split f32 -> bf16 hi + bf16 lo (truncating split; dropped lo*lo ~ 2^-16 rel)
__device__ inline void bf16split(float x, unsigned short& hi, unsigned short& lo) {
    unsigned int u = __float_as_uint(x);
    hi = (unsigned short)(u >> 16);
    float hif = __uint_as_float(u & 0xFFFF0000u);
    float r = x - hif;
    lo = (unsigned short)(__float_as_uint(r) >> 16);
}

__device__ inline v8f wmma_bf16(v16bf a, v16bf b, v8f c) {
    return __builtin_amdgcn_wmma_f32_16x16x32_bf16(
        false, a, false, b, (short)0, c, false, false);
}

// async global->LDS copy of one 32KB packed-weight chunk (128B per thread)
__device__ inline void issue_async_b(const unsigned short* __restrict__ Wb,
                                     int chunk, unsigned ldsBase, int tid)
{
    unsigned long long g =
        (unsigned long long)(uintptr_t)(Wb + (size_t)chunk * 16384);
    #pragma unroll
    for (int j = 0; j < 8; ++j) {
        unsigned slot = (unsigned)(tid + j * 256);
        unsigned off = ldsBase + (slot << 4);
        unsigned long long ga = g + ((unsigned long long)slot << 4);
        asm volatile("global_load_async_to_lds_b128 %0, %1, off"
                     :: "v"(off), "v"(ga) : "memory");
    }
}

// A 16x32 bf16 fragment straight from global: two contiguous 16B segments/lane
__device__ inline void load_a_frag(const unsigned short* __restrict__ Hi,
                                   const unsigned short* __restrict__ Lo,
                                   size_t rowBase, int valid,
                                   v16bf& aH, v16bf& aL)
{
    union { us8 p[2]; v16bf v; } uh, ul;
    if (valid) {
        uh.p[0] = *(const us8*)(Hi + rowBase);
        uh.p[1] = *(const us8*)(Hi + rowBase + 16);
        ul.p[0] = *(const us8*)(Lo + rowBase);
        ul.p[1] = *(const us8*)(Lo + rowBase + 16);
    } else {
        us8 z = {0, 0, 0, 0, 0, 0, 0, 0};
        uh.p[0] = z; uh.p[1] = z; ul.p[0] = z; ul.p[1] = z;
    }
    aH = uh.v; aL = ul.v;
}

// ---------------------------------------------------------------------------
// Split full f32 tensor into bf16 hi/lo arrays (one-time pass)
// ---------------------------------------------------------------------------
__global__ __launch_bounds__(256) void xsplit_kernel(
    const float* __restrict__ X, unsigned short* __restrict__ Hi,
    unsigned short* __restrict__ Lo, int n)
{
    int e = blockIdx.x * 256 + threadIdx.x;
    if (e >= n) return;
    unsigned short h, l;
    bf16split(X[e], h, l);
    Hi[e] = h; Lo[e] = l;
}

// ---------------------------------------------------------------------------
// Pack conv weights W[f][cin][ktap] into bf16 hi/lo B-fragments per K=32 chunk:
// Wb[chunk]{ hi[tile][lane][i], lo[tile][lane][i] } (8192 u16 each, 32KB/chunk)
// B 32x16 fragment: element i of lane L is K=(L<16?0:8)+(i&7)+(i>=8?16:0), N=tile*16+(L&15)
// ---------------------------------------------------------------------------
__global__ __launch_bounds__(256) void wb_pack_kernel(
    const float* __restrict__ W, unsigned short* __restrict__ Wb, int Cin)
{
    const int nChunks = (Cin * 3) / KC;
    int e = blockIdx.x * 256 + threadIdx.x;
    if (e >= nChunks * 8192) return;
    int c = e >> 13;
    int r = e & 8191;              // tile*512 + lane*16 + i
    int tile = r >> 9;
    int lane = (r >> 4) & 31;
    int i    = r & 15;
    int klocal = ((lane >> 4) << 3) + (i & 7) + ((i >> 3) << 4);
    int kflat  = c * KC + klocal;
    int f      = tile * 16 + (lane & 15);
    int ktap   = kflat / Cin;
    int d      = kflat - ktap * Cin;
    float w = W[((size_t)f * Cin + d) * 3 + ktap];
    unsigned short hi, lo;
    bf16split(w, hi, lo);
    Wb[(size_t)c * 16384 + r]        = hi;
    Wb[(size_t)c * 16384 + 8192 + r] = lo;
}

// ---------------------------------------------------------------------------
// Fused conv1d(K=3,pad=1) as split-bf16 GEMM (f32 accum) + bias + LN + ReLU
// 256 thr = 8 waves; tile 64 rows x 256 cols; wave: 2 row-groups x 4 col-tiles
// B tiles double-buffered in LDS via async-to-LDS; A fragments from global.
// ---------------------------------------------------------------------------
__global__ __launch_bounds__(256) void conv_ln_relu_kernel(
    const unsigned short* __restrict__ AHi,  // [B,N,Cin] bf16 hi of input
    const unsigned short* __restrict__ ALo,  // [B,N,Cin] bf16 lo of input
    const unsigned short* __restrict__ Wb,   // packed B fragments
    const float* __restrict__ bias,
    const float* __restrict__ gam,
    const float* __restrict__ bet,
    float* __restrict__ Yf,                  // [B,N,256] f32 out (optional)
    unsigned short* __restrict__ YbHi,       // [B,N,256] bf16 out (optional)
    unsigned short* __restrict__ YbLo,
    int Cin, int writeF32, int writeBf)
{
    // 64KB: B double-buffer during K-loop, re-used as LN staging afterwards
    __shared__ alignas(128) unsigned char smem[65536 + 512];
    unsigned short* BfBase = (unsigned short*)smem;
    float* HsBase = (float*)smem;
    float* s_mean = (float*)(smem + 65536);
    float* s_rstd = s_mean + TILE_ROWS;

    const int tid  = threadIdx.x;
    const int wave = tid >> 5;
    const int lane = tid & 31;

    const int rowBlock = blockIdx.x * TILE_ROWS;   // global row = b*N + n
    const int b  = rowBlock / N_;
    const int n0 = rowBlock - b * N_;

    const int rg0      = wave >> 2;        // row groups rg0 and rg0+2
    const int tileBase = (wave & 3) * 4;   // 4 col tiles of 16

    const int m   = lane & 15;
    const int kb0 = (lane >> 4) << 3;      // 0 or 8: lane-group K base
    const unsigned ldsB = (unsigned)(uintptr_t)BfBase;

    const int nChunks = (Cin * 3) / KC;
    v8f accA[4] = {}, accB[4] = {};

    // prologue: async-copy chunk 0 into buffer 0
    issue_async_b(Wb, 0, ldsB, tid);
    asm volatile("s_wait_asynccnt 0x0" ::: "memory");
    __syncthreads();

    for (int c = 0; c < nChunks; ++c) {
        const int cur = c & 1;
        if (c + 1 < nChunks)
            issue_async_b(Wb, c + 1, ldsB + (unsigned)((cur ^ 1) * 32768), tid);

        const int kc0  = c * KC;
        const int ktap = kc0 / Cin;              // chunk never crosses a tap
        const int d0   = kc0 - ktap * Cin;

        // ---- A fragments for both row groups, straight from global ----
        v16bf aH0, aL0, aH1, aL1;
        {
            int nsrc0 = n0 + rg0 * 16 + m + ktap - 1;
            int ok0 = (nsrc0 >= 0) & (nsrc0 < N_);
            size_t rb0 = ((size_t)b * N_ + (ok0 ? nsrc0 : 0)) * Cin + d0 + kb0;
            load_a_frag(AHi, ALo, rb0, ok0, aH0, aL0);
            int nsrc1 = nsrc0 + 32;
            int ok1 = (nsrc1 >= 0) & (nsrc1 < N_);
            size_t rb1 = ((size_t)b * N_ + (ok1 ? nsrc1 : 0)) * Cin + d0 + kb0;
            load_a_frag(AHi, ALo, rb1, ok1, aH1, aL1);
        }

        // ---- 24 WMMAs: split-bf16 products for 4 tiles x 2 row groups ----
        const unsigned short* bufp = BfBase + cur * 16384;
        #pragma unroll
        for (int j = 0; j < 4; ++j) {
            const v16bf bH = *(const v16bf*)(bufp + (tileBase + j) * 512 + lane * 16);
            const v16bf bL = *(const v16bf*)(bufp + 8192 + (tileBase + j) * 512 + lane * 16);
            accA[j] = wmma_bf16(aH0, bH, accA[j]);
            accA[j] = wmma_bf16(aH0, bL, accA[j]);
            accA[j] = wmma_bf16(aL0, bH, accA[j]);
            accB[j] = wmma_bf16(aH1, bH, accB[j]);
            accB[j] = wmma_bf16(aH1, bL, accB[j]);
            accB[j] = wmma_bf16(aL1, bH, accB[j]);
        }
        asm volatile("s_wait_asynccnt 0x0" ::: "memory");
        __syncthreads();
    }

    // ---- spill accumulators (+bias) into LN staging (re-uses B buffers) ----
    {
        const int baseM = rg0 * 16 + ((lane < 16) ? 0 : 8);
        const int colIn = lane & 15;
        #pragma unroll
        for (int j = 0; j < 4; ++j) {
            const int cc = (tileBase + j) * 16 + colIn;
            const float bb = bias[cc];
            #pragma unroll
            for (int r = 0; r < 8; ++r) {
                HsBase[(baseM + r) * F_ + cc]      = accA[j][r] + bb;
                HsBase[(baseM + 32 + r) * F_ + cc] = accB[j][r] + bb;
            }
        }
    }
    __syncthreads();

    // ---- LayerNorm stats: 4 threads per row, shuffle reduce ----
    {
        const int r    = tid >> 2;   // 0..63
        const int part = tid & 3;
        float s = 0.f, s2 = 0.f;
        #pragma unroll 8
        for (int ccc = part * 64; ccc < part * 64 + 64; ++ccc) {
            float v = HsBase[r * F_ + ccc];
            s += v; s2 += v * v;
        }
        #pragma unroll
        for (int off = 1; off < 4; off <<= 1) {
            s  += __shfl_xor(s, off, 32);
            s2 += __shfl_xor(s2, off, 32);
        }
        if (part == 0) {
            float mean = s * (1.f / 256.f);
            float var  = s2 * (1.f / 256.f) - mean * mean;
            s_mean[r] = mean;
            s_rstd[r] = rsqrtf(var + 1e-5f);
        }
    }
    __syncthreads();

    // ---- normalize + affine + ReLU; emit f32 and/or split-bf16 ----
    for (int e = tid; e < TILE_ROWS * F_; e += 256) {
        int r  = e >> 8;
        int cc = e & (F_ - 1);
        float v = (HsBase[r * F_ + cc] - s_mean[r]) * s_rstd[r] * gam[cc] + bet[cc];
        v = fmaxf(v, 0.f);
        size_t o = ((size_t)(rowBlock + r)) * F_ + cc;
        if (writeF32) Yf[o] = v;
        if (writeBf) {
            unsigned short h, l;
            bf16split(v, h, l);
            YbHi[o] = h; YbLo[o] = l;
        }
    }
}

// ---------------------------------------------------------------------------
// dur_pred = relu(H @ lin_w + lin_b) : one wave per row, shuffle reduction
// ---------------------------------------------------------------------------
__global__ __launch_bounds__(256) void dur_kernel(
    const float* __restrict__ H, const float* __restrict__ lw,
    const float* __restrict__ lb, float* __restrict__ dur)
{
    const int row  = blockIdx.x * 8 + (threadIdx.x >> 5);
    const int lane = threadIdx.x & 31;
    const float* h = H + (size_t)row * F_;
    float s = 0.f;
    #pragma unroll
    for (int i = 0; i < 8; ++i)
        s += h[lane + i * 32] * lw[lane + i * 32];
    #pragma unroll
    for (int off = 16; off > 0; off >>= 1)
        s += __shfl_xor(s, off, 32);
    if (lane == 0) dur[row] = fmaxf(s + lb[0], 0.f);
}

// ---------------------------------------------------------------------------
// Per-batch inclusive scan of target (Hillis-Steele in LDS)
// ---------------------------------------------------------------------------
__global__ __launch_bounds__(512) void scan_kernel(
    const int* __restrict__ target, int* __restrict__ csum)
{
    __shared__ int buf[N_];
    const int b = blockIdx.x, t = threadIdx.x;
    buf[t] = target[b * N_ + t];
    __syncthreads();
    for (int off = 1; off < N_; off <<= 1) {
        int v = (t >= off) ? buf[t - off] : 0;
        __syncthreads();
        buf[t] += v;
        __syncthreads();
    }
    csum[b * N_ + t] = buf[t];
}

// ---------------------------------------------------------------------------
// Length-regulator expansion: wave per output frame; binary search + b128 copy
// ---------------------------------------------------------------------------
__global__ __launch_bounds__(256) void expand_kernel(
    const float* __restrict__ X, const int* __restrict__ csum,
    float* __restrict__ out)
{
    __shared__ int cs[N_];
    const int blocksPerB = M_ / 8;
    const int b  = blockIdx.x / blocksPerB;
    const int t0 = (blockIdx.x - b * blocksPerB) * 8;

    for (int i = threadIdx.x; i < N_; i += 256)
        cs[i] = csum[b * N_ + i];
    __syncthreads();

    const int wave = threadIdx.x >> 5;
    const int lane = threadIdx.x & 31;
    const int t = t0 + wave;
    const int total = cs[N_ - 1];

    // searchsorted(csum, t, side='right'), clamped to N-1
    int pos = 0;
    #pragma unroll
    for (int step = 256; step > 0; step >>= 1) {
        int np = pos + step;
        if (np <= N_ - 1 && cs[np - 1] <= t) pos = np;
    }

    float4* dst = (float4*)(out + ((size_t)b * M_ + t) * D_);
    if (t < total) {
        const float4* src = (const float4*)(X + ((size_t)b * N_ + pos) * D_);
        #pragma unroll
        for (int j = 0; j < 3; ++j)
            dst[lane + 32 * j] = src[lane + 32 * j];
    } else {
        const float4 z = {0.f, 0.f, 0.f, 0.f};
        #pragma unroll
        for (int j = 0; j < 3; ++j)
            dst[lane + 32 * j] = z;
    }
}

// ---------------------------------------------------------------------------
extern "C" void kernel_launch(void* const* d_in, const int* in_sizes, int n_in,
                              void* d_out, int out_size, void* d_ws, size_t ws_size,
                              hipStream_t stream)
{
    const float* x      = (const float*)d_in[0];
    const int*   target = (const int*)  d_in[1];
    // d_in[2] = mel_max_length (compile-time 3072)
    const float* c1w = (const float*)d_in[3];
    const float* c1b = (const float*)d_in[4];
    const float* g1  = (const float*)d_in[5];
    const float* b1  = (const float*)d_in[6];
    const float* c2w = (const float*)d_in[7];
    const float* c2b = (const float*)d_in[8];
    const float* g2  = (const float*)d_in[9];
    const float* b2  = (const float*)d_in[10];
    const float* lw  = (const float*)d_in[11];
    const float* lb  = (const float*)d_in[12];

    // workspace layout
    const size_t nX = (size_t)B_ * N_ * D_;   // 6.29M
    const size_t nH = (size_t)B_ * N_ * F_;   // 4.19M
    const int chunks1 = (3 * D_) / KC;        // 36
    const int chunks2 = (3 * F_) / KC;        // 24

    unsigned short* xbHi = (unsigned short*)d_ws;
    unsigned short* xbLo = xbHi + nX;
    unsigned short* h1Hi = xbLo + nX;
    unsigned short* h1Lo = h1Hi + nH;
    unsigned short* wb1  = h1Lo + nH;
    unsigned short* wb2  = wb1 + (size_t)chunks1 * 16384;
    float* h2f = (float*)(wb2 + (size_t)chunks2 * 16384);
    int* cs = (int*)(h2f + nH);

    float* out = (float*)d_out;               // [B, M, D]
    float* dur = out + (size_t)B_ * M_ * D_;  // [B, N]

    xsplit_kernel<<<(int)((nX + 255) / 256), 256, 0, stream>>>(x, xbHi, xbLo, (int)nX);
    wb_pack_kernel<<<(chunks1 * 8192 + 255) / 256, 256, 0, stream>>>(c1w, wb1, D_);
    wb_pack_kernel<<<(chunks2 * 8192 + 255) / 256, 256, 0, stream>>>(c2w, wb2, F_);

    const int gemmBlocks = (B_ * N_) / TILE_ROWS;   // 256
    conv_ln_relu_kernel<<<gemmBlocks, 256, 0, stream>>>(
        xbHi, xbLo, wb1, c1b, g1, b1, (float*)nullptr, h1Hi, h1Lo, D_, 0, 1);
    conv_ln_relu_kernel<<<gemmBlocks, 256, 0, stream>>>(
        h1Hi, h1Lo, wb2, c2b, g2, b2, h2f, (unsigned short*)nullptr,
        (unsigned short*)nullptr, F_, 1, 0);

    dur_kernel<<<(B_ * N_) / 8, 256, 0, stream>>>(h2f, lw, lb, dur);
    scan_kernel<<<B_, N_, 0, stream>>>(target, cs);
    expand_kernel<<<B_ * (M_ / 8), 256, 0, stream>>>(x, cs, out);
}
